// MultiHeadAttention_57088705299043
// MI455X (gfx1250) — matrix operand order repaired and checked by reading, not compiled, not run on hardware
//
#include <hip/hip_runtime.h>
#include <hip/hip_bf16.h>
#include <math.h>

typedef __attribute__((ext_vector_type(16))) _Float16 v16h;
typedef __attribute__((ext_vector_type(8)))  _Float16 v8h;
typedef __attribute__((ext_vector_type(8)))  float    v8f;

#define DEV __device__ __forceinline__

constexpr int MODEL_DIM = 768;
constexpr int HEADS     = 12;
constexpr int HEAD_DIM  = 64;
constexpr int SEQ       = 2048;
constexpr int BATCH     = 4;
constexpr int ROWS      = BATCH * SEQ;      // 8192
constexpr int MT        = ROWS / 16;        // 512
constexpr int NT        = MODEL_DIM / 16;   // 48
constexpr int MT2       = MT / 2;           // 256  (32-row blocks)
constexpr int NT4       = NT / 4;           // 12   (64-col blocks == heads)

// ---------- WMMA fragment helpers (wave32, f16 16x16x32) ----------
// A fragment (16x32 MxK): lane L -> M = L%16; halves j: K = g*16 + j, g=L/16
DEV v16h loadA16(const _Float16* p, int ld, int ln, int g, int k0) {
  const v8h* p0 = (const v8h*)(p + (size_t)ln * ld + k0 + g * 16);
  const v8h* p1 = (const v8h*)(p + (size_t)ln * ld + k0 + g * 16 + 8);
  v8h lo = *p0;
  v8h hi = *p1;
  return __builtin_shufflevector(lo, hi, 0,1,2,3,4,5,6,7,8,9,10,11,12,13,14,15);
}
// B fragment (32x16 KxN) from [N][K] row-major storage: lane L -> N = L%16; K = g*16 + j
DEV v16h loadB16(const _Float16* p, int ld, int ln, int g, int k0) {
  const v8h* q = (const v8h*)(p + (size_t)ln * ld + k0 + g * 16);
  v8h lo = q[0];
  v8h hi = q[1];
  return __builtin_shufflevector(lo, hi, 0,1,2,3,4,5,6,7,8,9,10,11,12,13,14,15);
}
DEV v8f wmma_f16(v16h a, v16h b, v8f c) {
  return __builtin_amdgcn_wmma_f32_16x16x32_f16(false, a, false, b, (short)0, c, false, false);
}

// ---------- fp32 -> f16 convert ----------
__global__ void k_cvt(const float* __restrict__ s, _Float16* __restrict__ d, int n) {
  int i = blockIdx.x * blockDim.x + threadIdx.x;
  if (i < n) d[i] = (_Float16)s[i];
}

// ---------- QKV projection + RoPE + layout transform (2Mx4N register blocking) ----------
__global__ __launch_bounds__(256) void k_qkv(const _Float16* __restrict__ xh,
                                             const _Float16* __restrict__ wqkv,
                                             _Float16* __restrict__ Qh,
                                             _Float16* __restrict__ Kh,
                                             _Float16* __restrict__ Vt) {
  int wv   = blockIdx.x * (blockDim.x >> 5) + (threadIdx.x >> 5);
  int lane = threadIdx.x & 31;
  int g = lane >> 4, ln = lane & 15;
  int mat = wv / (MT2 * NT4);
  int rem = wv % (MT2 * NT4);
  int mt = rem / NT4, nt = rem % NT4;

  const _Float16* A0 = xh + (size_t)mt * 32 * MODEL_DIM;
  const _Float16* A1 = A0 + 16 * MODEL_DIM;
  const _Float16* B0 = wqkv + (size_t)mat * MODEL_DIM * MODEL_DIM + (size_t)nt * 64 * MODEL_DIM;

  v8f acc[2][4] = {};
#pragma unroll 2
  for (int k0 = 0; k0 < MODEL_DIM; k0 += 32) {
    v16h a0 = loadA16(A0, MODEL_DIM, ln, g, k0);
    v16h a1 = loadA16(A1, MODEL_DIM, ln, g, k0);
    v16h b0 = loadB16(B0,                  MODEL_DIM, ln, g, k0);
    v16h b1 = loadB16(B0 + 16 * MODEL_DIM, MODEL_DIM, ln, g, k0);
    v16h b2 = loadB16(B0 + 32 * MODEL_DIM, MODEL_DIM, ln, g, k0);
    v16h b3 = loadB16(B0 + 48 * MODEL_DIM, MODEL_DIM, ln, g, k0);
    acc[0][0] = wmma_f16(a0, b0, acc[0][0]);
    acc[0][1] = wmma_f16(a0, b1, acc[0][1]);
    acc[0][2] = wmma_f16(a0, b2, acc[0][2]);
    acc[0][3] = wmma_f16(a0, b3, acc[0][3]);
    acc[1][0] = wmma_f16(a1, b0, acc[1][0]);
    acc[1][1] = wmma_f16(a1, b1, acc[1][1]);
    acc[1][2] = wmma_f16(a1, b2, acc[1][2]);
    acc[1][3] = wmma_f16(a1, b3, acc[1][3]);
  }

  int hh = nt;  // 64-col block == head index
  if (mat == 2) {
#pragma unroll
    for (int ni = 0; ni < 4; ni++) {
      int dd = ni * 16 + ln;
#pragma unroll
      for (int mi = 0; mi < 2; mi++) {
#pragma unroll
        for (int r = 0; r < 8; r++) {
          int R = mt * 32 + mi * 16 + r + 8 * g;
          int bb = R >> 11, t = R & (SEQ - 1);
          Vt[(((size_t)bb * HEADS + hh) * HEAD_DIM + dd) * SEQ + t] = (_Float16)acc[mi][ni][r];
        }
      }
    }
  } else {
    _Float16* dst = (mat == 0) ? Qh : Kh;
#pragma unroll
    for (int ni = 0; ni < 4; ni++) {
      int Cc = nt * 64 + ni * 16 + ln;
      int dd = ni * 16 + ln;
      float theta = __expf((float)(Cc & ~1) * (-9.210340371976184f / 768.f));
#pragma unroll
      for (int mi = 0; mi < 2; mi++) {
#pragma unroll
        for (int r = 0; r < 8; r++) {
          float v  = acc[mi][ni][r];
          float pv = __shfl_xor(v, 1);   // rotation partner column (Cc^1) lives in lane^1
          int R = mt * 32 + mi * 16 + r + 8 * g;
          int bb = R >> 11, t = R & (SEQ - 1);
          float ang = (float)(t + 1) * theta;
          float sn, cs;
          __sincosf(ang, &sn, &cs);
          float o = (Cc & 1) ? (v * cs + pv * sn) : (v * cs - pv * sn);
          dst[(((size_t)bb * HEADS + hh) * SEQ + t) * HEAD_DIM + dd] = (_Float16)o;
        }
      }
    }
  }
}

// ---------- online softmax tile update (D-layout scores -> LDS P tile) ----------
DEV void online_update(const v8f& sa, const v8f& sb, bool hb,
                       int qrow0, int s0, int ln, int g,
                       float* m, float* l, v8f* o, _Float16* lp) {
#pragma unroll
  for (int r = 0; r < 8; r++) {
    int qm = qrow0 + r + 8 * g;
    float va = (s0 + ln <= qm) ? sa[r] * 0.125f : -__builtin_inff();
    float vb = (hb && (s0 + 16 + ln <= qm)) ? sb[r] * 0.125f : -__builtin_inff();
    float rm = fmaxf(va, vb);
    rm = fmaxf(rm, __shfl_xor(rm, 1));
    rm = fmaxf(rm, __shfl_xor(rm, 2));
    rm = fmaxf(rm, __shfl_xor(rm, 4));
    rm = fmaxf(rm, __shfl_xor(rm, 8));
    float mnew  = fmaxf(m[r], rm);
    float alpha = __expf(m[r] - mnew);
    float pa = __expf(va - mnew);
    float pb = __expf(vb - mnew);
    float rs = pa + pb;
    rs += __shfl_xor(rs, 1);
    rs += __shfl_xor(rs, 2);
    rs += __shfl_xor(rs, 4);
    rs += __shfl_xor(rs, 8);
    l[r] = l[r] * alpha + rs;
    m[r] = mnew;
    o[0][r] *= alpha; o[1][r] *= alpha; o[2][r] *= alpha; o[3][r] *= alpha;
    lp[(r + 8 * g) * 32 + ln]      = (_Float16)pa;
    lp[(r + 8 * g) * 32 + 16 + ln] = (_Float16)pb;
  }
}

// ---------- flash-style causal attention: one wave per (b,h,32-query tile) ----------
__global__ __launch_bounds__(128) void k_attn(const _Float16* __restrict__ Qh,
                                              const _Float16* __restrict__ Kh,
                                              const _Float16* __restrict__ Vt,
                                              _Float16* __restrict__ Oh) {
  __shared__ __align__(16) _Float16 ldsP[4][2][16 * 32];
  int w = threadIdx.x >> 5;
  int lane = threadIdx.x & 31;
  int g = lane >> 4, ln = lane & 15;
  int wid = blockIdx.x * 4 + w;
  int qt = wid & (SEQ / 32 - 1);
  int bh = wid >> 6;
  int bb = bh / HEADS, h = bh % HEADS;
  int q0 = qt * 32;

  const _Float16* qp  = Qh + ((size_t)bh * SEQ + q0) * HEAD_DIM;
  const _Float16* kp0 = Kh + (size_t)bh * SEQ * HEAD_DIM;
  const _Float16* vp  = Vt + (size_t)bh * HEAD_DIM * SEQ;

  v16h aqA0 = loadA16(qp, HEAD_DIM, ln, g, 0);
  v16h aqA1 = loadA16(qp, HEAD_DIM, ln, g, 32);
  v16h aqB0 = loadA16(qp + 16 * HEAD_DIM, HEAD_DIM, ln, g, 0);
  v16h aqB1 = loadA16(qp + 16 * HEAD_DIM, HEAD_DIM, ln, g, 32);

  v8f oA[4] = {}, oB[4] = {};
  float mA[8], lA[8], mB[8], lB[8];
#pragma unroll
  for (int r = 0; r < 8; r++) {
    mA[r] = -__builtin_inff(); lA[r] = 0.f;
    mB[r] = -__builtin_inff(); lB[r] = 0.f;
  }

  int qlastA = q0 + 15, qlastB = q0 + 31;
  for (int s0 = 0; s0 <= qlastB; s0 += 32) {
    const _Float16* kp = kp0 + (size_t)s0 * HEAD_DIM;
    bool doA = (s0 <= qlastA);
    bool hbA = ((s0 + 16) <= qlastA);

    v16h bk00 = loadB16(kp, HEAD_DIM, ln, g, 0);
    v16h bk01 = loadB16(kp, HEAD_DIM, ln, g, 32);
    v16h bk10 = loadB16(kp + 16 * HEAD_DIM, HEAD_DIM, ln, g, 0);
    v16h bk11 = loadB16(kp + 16 * HEAD_DIM, HEAD_DIM, ln, g, 32);
    __builtin_prefetch(kp + 32 * HEAD_DIM, 0, 3);          // next K chunk
    __builtin_prefetch(vp + s0 + 32, 0, 3);                // next V chunk (row 0)

    v8f sBa = {}, sBb = {};
    sBa = wmma_f16(aqB0, bk00, sBa);
    sBa = wmma_f16(aqB1, bk01, sBa);
    sBb = wmma_f16(aqB0, bk10, sBb);
    sBb = wmma_f16(aqB1, bk11, sBb);

    if (doA) {
      v8f sAa = {}, sAb = {};
      sAa = wmma_f16(aqA0, bk00, sAa);
      sAa = wmma_f16(aqA1, bk01, sAa);
      if (hbA) {
        sAb = wmma_f16(aqA0, bk10, sAb);
        sAb = wmma_f16(aqA1, bk11, sAb);
      }
      online_update(sAa, sAb, hbA, q0, s0, ln, g, mA, lA, oA, &ldsP[w][0][0]);
    }
    online_update(sBa, sBb, true, q0 + 16, s0, ln, g, mB, lB, oB, &ldsP[w][1][0]);

    asm volatile("s_wait_dscnt 0x0" ::: "memory");          // same-wave LDS RAW

    v16h bv0 = loadB16(vp + (size_t)0 * 16 * SEQ, SEQ, ln, g, s0);
    v16h bv1 = loadB16(vp + (size_t)1 * 16 * SEQ, SEQ, ln, g, s0);
    v16h bv2 = loadB16(vp + (size_t)2 * 16 * SEQ, SEQ, ln, g, s0);
    v16h bv3 = loadB16(vp + (size_t)3 * 16 * SEQ, SEQ, ln, g, s0);

    v16h pfB = loadA16(&ldsP[w][1][0], 32, ln, g, 0);
    oB[0] = wmma_f16(pfB, bv0, oB[0]);
    oB[1] = wmma_f16(pfB, bv1, oB[1]);
    oB[2] = wmma_f16(pfB, bv2, oB[2]);
    oB[3] = wmma_f16(pfB, bv3, oB[3]);
    if (doA) {
      v16h pfA = loadA16(&ldsP[w][0][0], 32, ln, g, 0);
      oA[0] = wmma_f16(pfA, bv0, oA[0]);
      oA[1] = wmma_f16(pfA, bv1, oA[1]);
      oA[2] = wmma_f16(pfA, bv2, oA[2]);
      oA[3] = wmma_f16(pfA, bv3, oA[3]);
    }
    asm volatile("" ::: "memory");
  }

#pragma unroll
  for (int r = 0; r < 8; r++) {
    float invA = 1.f / lA[r];
    float invB = 1.f / lB[r];
    int tA = q0 + r + 8 * g;
    int tB = q0 + 16 + r + 8 * g;
    _Float16* rowA = Oh + ((size_t)bb * SEQ + tA) * MODEL_DIM + h * HEAD_DIM + ln;
    _Float16* rowB = Oh + ((size_t)bb * SEQ + tB) * MODEL_DIM + h * HEAD_DIM + ln;
    rowA[0]  = (_Float16)(oA[0][r] * invA);
    rowA[16] = (_Float16)(oA[1][r] * invA);
    rowA[32] = (_Float16)(oA[2][r] * invA);
    rowA[48] = (_Float16)(oA[3][r] * invA);
    rowB[0]  = (_Float16)(oB[0][r] * invB);
    rowB[16] = (_Float16)(oB[1][r] * invB);
    rowB[32] = (_Float16)(oB[2][r] * invB);
    rowB[48] = (_Float16)(oB[3][r] * invB);
  }
}

// ---------- output projection: out = O @ Wo^T (fp32 out), 2Mx4N blocking ----------
__global__ __launch_bounds__(256) void k_oproj(const _Float16* __restrict__ Oh,
                                               const _Float16* __restrict__ wo,
                                               float* __restrict__ out) {
  int wv   = blockIdx.x * (blockDim.x >> 5) + (threadIdx.x >> 5);
  int lane = threadIdx.x & 31;
  int g = lane >> 4, ln = lane & 15;
  int mt = wv / NT4, nt = wv % NT4;

  const _Float16* A0 = Oh + (size_t)mt * 32 * MODEL_DIM;
  const _Float16* A1 = A0 + 16 * MODEL_DIM;
  const _Float16* B0 = wo + (size_t)nt * 64 * MODEL_DIM;

  v8f acc[2][4] = {};
#pragma unroll 2
  for (int k0 = 0; k0 < MODEL_DIM; k0 += 32) {
    v16h a0 = loadA16(A0, MODEL_DIM, ln, g, k0);
    v16h a1 = loadA16(A1, MODEL_DIM, ln, g, k0);
    v16h b0 = loadB16(B0,                  MODEL_DIM, ln, g, k0);
    v16h b1 = loadB16(B0 + 16 * MODEL_DIM, MODEL_DIM, ln, g, k0);
    v16h b2 = loadB16(B0 + 32 * MODEL_DIM, MODEL_DIM, ln, g, k0);
    v16h b3 = loadB16(B0 + 48 * MODEL_DIM, MODEL_DIM, ln, g, k0);
    acc[0][0] = wmma_f16(a0, b0, acc[0][0]);
    acc[0][1] = wmma_f16(a0, b1, acc[0][1]);
    acc[0][2] = wmma_f16(a0, b2, acc[0][2]);
    acc[0][3] = wmma_f16(a0, b3, acc[0][3]);
    acc[1][0] = wmma_f16(a1, b0, acc[1][0]);
    acc[1][1] = wmma_f16(a1, b1, acc[1][1]);
    acc[1][2] = wmma_f16(a1, b2, acc[1][2]);
    acc[1][3] = wmma_f16(a1, b3, acc[1][3]);
  }
#pragma unroll
  for (int ni = 0; ni < 4; ni++) {
    int Cc = nt * 64 + ni * 16 + ln;
#pragma unroll
    for (int mi = 0; mi < 2; mi++) {
#pragma unroll
      for (int r = 0; r < 8; r++) {
        int R = mt * 32 + mi * 16 + r + 8 * g;
        out[(size_t)R * MODEL_DIM + Cc] = acc[mi][ni][r];
      }
    }
  }
}

extern "C" void kernel_launch(void* const* d_in, const int* in_sizes, int n_in,
                              void* d_out, int out_size, void* d_ws, size_t ws_size,
                              hipStream_t stream) {
  const float* x  = (const float*)d_in[0];
  const float* Wq = (const float*)d_in[1];
  const float* Wk = (const float*)d_in[2];
  const float* Wv = (const float*)d_in[3];
  const float* Wo = (const float*)d_in[4];
  float* outF = (float*)d_out;

  const size_t nX = (size_t)ROWS * MODEL_DIM;        // 6,291,456
  const size_t nW = (size_t)MODEL_DIM * MODEL_DIM;   // 589,824

  char* ws = (char*)d_ws;
  _Float16* xh   = (_Float16*)ws; ws += nX * 2;
  _Float16* w16  = (_Float16*)ws; ws += 3 * nW * 2;  // Wq,Wk,Wv packed
  _Float16* wo16 = (_Float16*)ws; ws += nW * 2;
  _Float16* Qh   = (_Float16*)ws; ws += nX * 2;      // [b][h][t][64]
  _Float16* Kh   = (_Float16*)ws; ws += nX * 2;      // [b][h][t][64]
  _Float16* Vt   = (_Float16*)ws; ws += nX * 2;      // [b][h][64][t]
  _Float16* Oh   = (_Float16*)ws; ws += nX * 2;      // [b][t][768]

  k_cvt<<<(int)((nX + 255) / 256), 256, 0, stream>>>(x, xh, (int)nX);
  k_cvt<<<(int)((nW + 255) / 256), 256, 0, stream>>>(Wq, w16, (int)nW);
  k_cvt<<<(int)((nW + 255) / 256), 256, 0, stream>>>(Wk, w16 + nW, (int)nW);
  k_cvt<<<(int)((nW + 255) / 256), 256, 0, stream>>>(Wv, w16 + 2 * nW, (int)nW);
  k_cvt<<<(int)((nW + 255) / 256), 256, 0, stream>>>(Wo, wo16, (int)nW);

  k_qkv<<<(3 * MT2 * NT4) / 8, 256, 0, stream>>>(xh, w16, Qh, Kh, Vt);           // 1152 blocks
  k_attn<<<(BATCH * HEADS * (SEQ / 32)) / 4, 128, 0, stream>>>(Qh, Kh, Vt, Oh);  // 768 blocks
  k_oproj<<<(MT2 * NT4) / 8, 256, 0, stream>>>(Oh, wo16, outF);                  // 384 blocks
}
